// SlotAttention_49100066127986
// MI455X (gfx1250) — matrix operand order repaired and checked by reading, not compiled, not run on hardware
//
#include <hip/hip_runtime.h>
#include <hip/hip_bf16.h>

typedef __attribute__((ext_vector_type(16))) __bf16 v16bf;
typedef __attribute__((ext_vector_type(8)))  __bf16 v8bf;
typedef __attribute__((ext_vector_type(8)))  float  v8f;

#define WMMA_BF16(a, b, c) \
  __builtin_amdgcn_wmma_f32_16x16x32_bf16(false, (a), false, (b), (short)0, (c), false, false)

static __device__ __forceinline__ v16bf cat16(v8bf lo, v8bf hi) {
  return __builtin_shufflevector(lo, hi, 0,1,2,3,4,5,6,7,8,9,10,11,12,13,14,15);
}

constexpr int Bb = 64;       // batch
constexpr int Nn = 4096;     // tokens
constexpr int Dd = 128;      // feature dim
constexpr int Kk = 8;        // slots

// ---------------------------------------------------------------------------
// Kernel A: LN(x) then keys = xn @ Wk^T + bk  (stored [b][n][d] bf16)
//                     values = xn @ Wv^T + bv (stored TRANSPOSED [b][d][n] bf16)
// ---------------------------------------------------------------------------
__global__ __launch_bounds__(256) void ln_kv_kernel(
    const float* __restrict__ x,
    const float* __restrict__ Wk, const float* __restrict__ bk,
    const float* __restrict__ Wv, const float* __restrict__ bv,
    const float* __restrict__ g_in, const float* __restrict__ b_in,
    __bf16* __restrict__ keys, __bf16* __restrict__ valsT)
{
  __shared__ __bf16 xn_s[64 * 128];     // 16 KB
  __shared__ __bf16 w_s[128 * 128];     // 32 KB (reused for Wk then Wv)
  __shared__ __bf16 st_s[8][16 * 16];   //  4 KB, per-wave C-tile staging

  const int tid  = threadIdx.x;
  const int lane = tid & 31;
  const int wave = tid >> 5;
  const int b    = blockIdx.y;
  const int tok0 = blockIdx.x * 64;

  const int kL = lane * 4;
  const float4 gv  = *(const float4*)(g_in + kL);
  const float4 bv4 = *(const float4*)(b_in + kL);

  // ---- LayerNorm: each wave handles 8 rows, 4 elements per lane ----
  for (int rr = 0; rr < 8; ++rr) {
    int row = wave * 8 + rr;
    const float* xr = x + ((size_t)b * Nn + tok0 + row) * Dd;
    float4 v = *(const float4*)(xr + kL);
    float s = v.x + v.y + v.z + v.w;
    for (int m = 16; m >= 1; m >>= 1) s += __shfl_xor(s, m, 32);
    float mean = s * (1.0f / 128.0f);
    float dx = v.x - mean, dy = v.y - mean, dz = v.z - mean, dw = v.w - mean;
    float q = dx*dx + dy*dy + dz*dz + dw*dw;
    for (int m = 16; m >= 1; m >>= 1) q += __shfl_xor(q, m, 32);
    float rinv = rsqrtf(q * (1.0f / 128.0f) + 1e-5f);
    xn_s[row*128 + kL+0] = (__bf16)(dx * rinv * gv.x + bv4.x);
    xn_s[row*128 + kL+1] = (__bf16)(dy * rinv * gv.y + bv4.y);
    xn_s[row*128 + kL+2] = (__bf16)(dz * rinv * gv.z + bv4.z);
    xn_s[row*128 + kL+3] = (__bf16)(dw * rinv * gv.w + bv4.w);
  }
  __syncthreads();

  const int mtile = wave & 3;
  const int nt0   = (wave >> 2) * 4;
  const int row_a = lane & 15;
  const int half  = lane >> 4;
  const int arow  = mtile * 16 + row_a;
  v16bf af[4];
  #pragma unroll
  for (int ks = 0; ks < 4; ++ks) {
    int kb = ks * 32 + half * 16;
    af[ks] = cat16(*(const v8bf*)&xn_s[arow * 128 + kb],
                   *(const v8bf*)&xn_s[arow * 128 + kb + 8]);
  }

  #pragma unroll
  for (int sel = 0; sel < 2; ++sel) {     // unrolled: sel is compile-time
    __syncthreads();
    const float* W = sel ? Wv : Wk;
    for (int i = tid; i < 128 * 128; i += 256) w_s[i] = (__bf16)W[i];
    __syncthreads();
    const float* bias = sel ? bv : bk;

    for (int t = 0; t < 4; ++t) {
      int nt = nt0 + t;
      v8f acc = {0.f,0.f,0.f,0.f,0.f,0.f,0.f,0.f};
      #pragma unroll
      for (int ks = 0; ks < 4; ++ks) {
        int n  = nt * 16 + row_a;
        int ko = ks * 32 + half * 16;
        acc = WMMA_BF16(af[ks],
                        cat16(*(const v8bf*)&w_s[n * 128 + ko],
                              *(const v8bf*)&w_s[n * 128 + ko + 8]),
                        acc);
      }
      int dcol = nt * 16 + row_a;
      float bb = bias[dcol];
      if (sel == 1) {
        v8bf pk;
        #pragma unroll
        for (int e = 0; e < 8; ++e) pk[e] = (__bf16)(acc[e] + bb);
        int t0 = tok0 + mtile * 16 + half * 8;
        *(v8bf*)&valsT[((size_t)b * Dd + dcol) * Nn + t0] = pk;
      } else {
        #pragma unroll
        for (int e = 0; e < 8; ++e)
          st_s[wave][(half * 8 + e) * 16 + row_a] = (__bf16)(acc[e] + bb);
        int r  = lane >> 1;
        int hh = lane & 1;
        v8bf pk = *(const v8bf*)&st_s[wave][r * 16 + hh * 8];
        int token = tok0 + mtile * 16 + r;
        *(v8bf*)&keys[((size_t)b * Nn + token) * Dd + nt * 16 + hh * 8] = pk;
      }
    }
  }
}

// ---------------------------------------------------------------------------
__global__ void zero_kernel(float* p, int n) {
  int i = blockIdx.x * blockDim.x + threadIdx.x;
  if (i < n) p[i] = 0.0f;
}
__global__ void cvtbf_kernel(const float* __restrict__ src, __bf16* __restrict__ dst, int n) {
  int i = blockIdx.x * blockDim.x + threadIdx.x;
  if (i < n) dst[i] = (__bf16)src[i];
}

// ---------------------------------------------------------------------------
// Kernel C: one (batch, 512-token chunk) per block. All matmuls WMMA.
// ---------------------------------------------------------------------------
__global__ __launch_bounds__(256) void attn_kernel(
    const float* __restrict__ slots,
    const __bf16* __restrict__ keys, const __bf16* __restrict__ valsT,
    const __bf16* __restrict__ Wq_bf, const float* __restrict__ bq,
    const float* __restrict__ g_slot, const float* __restrict__ b_slot,
    float* __restrict__ updp, float* __restrict__ rowsum)
{
  __shared__ __bf16 s_bf[16 * 128];      // 4 KB  LN(slots), rows 8..15 zero
  __shared__ __bf16 q_s[16 * 128];       // 4 KB
  __shared__ __bf16 attn_s[8][16 * 32];  // 8 KB per-wave scratch
  __shared__ float  upd_s[16 * 128];     // 8 KB block accumulator
  __shared__ float  rs_s[16];

  const int tid  = threadIdx.x;
  const int lane = tid & 31;
  const int wave = tid >> 5;
  const int row_a = lane & 15;
  const int half  = lane >> 4;
  const int b     = blockIdx.y;
  const int chunk = blockIdx.x;

  for (int t = tid; t < 16 * 128; t += 256) upd_s[t] = 0.f;
  if (tid < 16) rs_s[tid] = 0.f;
  #pragma unroll
  for (int t = 0; t < 4; ++t) s_bf[1024 + tid * 4 + t] = (__bf16)0.0f;

  // LN(slots): wave w handles slot row w, straight from global
  {
    int row = wave;
    const float* sr = slots + (size_t)b * Kk * Dd + row * 128;
    float4 v = *(const float4*)(sr + lane * 4);
    float s = v.x + v.y + v.z + v.w;
    for (int m = 16; m >= 1; m >>= 1) s += __shfl_xor(s, m, 32);
    float mean = s * (1.0f / 128.0f);
    float dx = v.x - mean, dy = v.y - mean, dz = v.z - mean, dw = v.w - mean;
    float q = dx*dx + dy*dy + dz*dz + dw*dw;
    for (int m = 16; m >= 1; m >>= 1) q += __shfl_xor(q, m, 32);
    float rinv = rsqrtf(q * (1.0f / 128.0f) + 1e-5f);
    int k = lane * 4;
    s_bf[row*128 + k+0] = (__bf16)(dx * rinv * g_slot[k+0] + b_slot[k+0]);
    s_bf[row*128 + k+1] = (__bf16)(dy * rinv * g_slot[k+1] + b_slot[k+1]);
    s_bf[row*128 + k+2] = (__bf16)(dz * rinv * g_slot[k+2] + b_slot[k+2]);
    s_bf[row*128 + k+3] = (__bf16)(dw * rinv * g_slot[k+3] + b_slot[k+3]);
  }
  __syncthreads();

  // q = s @ Wq^T + bq via WMMA; wave computes column tile nt = wave.
  // (q padding rows only affect discarded logits rows -> bias everywhere OK)
  {
    v16bf as4[4];
    #pragma unroll
    for (int ks = 0; ks < 4; ++ks) {
      int kb = ks * 32 + half * 16;
      as4[ks] = cat16(*(const v8bf*)&s_bf[row_a * 128 + kb],
                      *(const v8bf*)&s_bf[row_a * 128 + kb + 8]);
    }
    int dcol = wave * 16 + row_a;
    v8f acc = {0.f,0.f,0.f,0.f,0.f,0.f,0.f,0.f};
    #pragma unroll
    for (int ks = 0; ks < 4; ++ks) {
      int ko = ks * 32 + half * 16;
      acc = WMMA_BF16(as4[ks],
                      cat16(*(const v8bf*)&Wq_bf[dcol * 128 + ko],
                            *(const v8bf*)&Wq_bf[dcol * 128 + ko + 8]),
                      acc);
    }
    float bb = bq[dcol];
    #pragma unroll
    for (int e = 0; e < 8; ++e)
      q_s[(half * 8 + e) * 128 + dcol] = (__bf16)(acc[e] + bb);
  }
  __syncthreads();

  v16bf qf[4];
  #pragma unroll
  for (int ks = 0; ks < 4; ++ks) {
    int kb = ks * 32 + half * 16;
    qf[ks] = cat16(*(const v8bf*)&q_s[row_a * 128 + kb],
                   *(const v8bf*)&q_s[row_a * 128 + kb + 8]);
  }

  const float scale = 0.08838834764831845f;  // D^-0.5
  v8f upd[8];
  #pragma unroll
  for (int dt = 0; dt < 8; ++dt) {
    v8f z = {0.f,0.f,0.f,0.f,0.f,0.f,0.f,0.f};
    upd[dt] = z;
  }
  float rsum[8] = {0.f,0.f,0.f,0.f,0.f,0.f,0.f,0.f};

  const int tokbase = chunk * 512 + wave * 64;
  __bf16* aw = attn_s[wave];

  for (int g = 0; g < 2; ++g) {
    int tg = tokbase + g * 32;
    __builtin_prefetch(keys + ((size_t)b * Nn + tg + 32) * Dd, 0, 1);

    #pragma unroll
    for (int tt = 0; tt < 2; ++tt) {
      v8f c = {0.f,0.f,0.f,0.f,0.f,0.f,0.f,0.f};
      int token = tg + tt * 16 + row_a;
      const __bf16* kp = keys + ((size_t)b * Nn + token) * Dd + half * 16;
      #pragma unroll
      for (int ks = 0; ks < 4; ++ks) {
        c = WMMA_BF16(qf[ks],
                      cat16(*(const v8bf*)(kp + ks * 32),
                            *(const v8bf*)(kp + ks * 32 + 8)),
                      c);
      }
      if (lane < 16) {   // softmax over SLOT axis: per-lane over 8 C-VGPRs
        float m = -3.4e38f;
        #pragma unroll
        for (int e = 0; e < 8; ++e) { c[e] *= scale; m = fmaxf(m, c[e]); }
        float ssum = 0.f;
        #pragma unroll
        for (int e = 0; e < 8; ++e) { c[e] = __expf(c[e] - m); ssum += c[e]; }
        float inv = 1.0f / ssum;
        #pragma unroll
        for (int e = 0; e < 8; ++e) { c[e] = c[e] * inv + 1e-8f; rsum[e] += c[e]; }
      } else {
        #pragma unroll
        for (int e = 0; e < 8; ++e) c[e] = 0.f;
      }
      int col = tt * 16 + row_a;
      #pragma unroll
      for (int e = 0; e < 8; ++e)
        aw[(half * 8 + e) * 32 + col] = (__bf16)c[e];
    }

    v16bf afr = cat16(*(const v8bf*)&aw[row_a * 32 + half * 16],
                      *(const v8bf*)&aw[row_a * 32 + half * 16 + 8]);
    #pragma unroll
    for (int dt = 0; dt < 8; ++dt) {
      const __bf16* vp = valsT + ((size_t)b * Dd + dt * 16 + row_a) * Nn + tg + half * 16;
      upd[dt] = WMMA_BF16(afr,
                          cat16(*(const v8bf*)vp, *(const v8bf*)(vp + 8)),
                          upd[dt]);
    }
  }

  // block-level reduction in LDS (ds_add_f32), then single global flush
  #pragma unroll
  for (int dt = 0; dt < 8; ++dt) {
    #pragma unroll
    for (int e = 0; e < 8; ++e)
      atomicAdd(&upd_s[(half * 8 + e) * 128 + dt * 16 + row_a], upd[dt][e]);
  }
  #pragma unroll
  for (int e = 0; e < 8; ++e) {
    float v = rsum[e];
    for (int m = 1; m <= 8; m <<= 1) v += __shfl_xor(v, m, 32);
    if (lane == 0) atomicAdd(&rs_s[e], v);
  }
  __syncthreads();

  for (int t = tid; t < 8 * 128; t += 256)
    atomicAdd(&updp[(size_t)b * 16 * Dd + t], upd_s[t]);
  if (tid < 8) atomicAdd(&rowsum[b * 16 + tid], rs_s[tid]);
}

// ---------------------------------------------------------------------------
// Kernel D: normalize updates, GRU cell, LN + MLP — all matmuls WMMA.
// One block per batch. Padding rows only pollute discarded C rows 8..15.
// ---------------------------------------------------------------------------
__global__ __launch_bounds__(256) void gru_mlp_kernel(
    float* __restrict__ slots,
    const float* __restrict__ updp, const float* __restrict__ rowsum,
    const __bf16* __restrict__ Wih_bf, const __bf16* __restrict__ Whh_bf,
    const float* __restrict__ b_ih, const float* __restrict__ b_hh,
    const float* __restrict__ g_mlp, const float* __restrict__ b_mlp,
    const __bf16* __restrict__ W1_bf, const float* __restrict__ b1,
    const __bf16* __restrict__ W2_bf, const float* __restrict__ b2)
{
  __shared__ __bf16 u_bf[16*128], h_bf[16*128];      // 4 KB each (A matrices)
  __shared__ __bf16 ln_bf[16*128], mid_bf[16*128];   // 4 KB each
  __shared__ float  h_s[8*128], hn_s[8*128];         // 4 KB each
  __shared__ float  gi_s[8*384], gh_s[8*384];        // 12 KB each  (48 KB total)

  const int tid  = threadIdx.x;
  const int lane = tid & 31;
  const int wave = tid >> 5;
  const int row_a = lane & 15;
  const int half  = lane >> 4;
  const int b    = blockIdx.x;

  #pragma unroll
  for (int t = 0; t < 4; ++t) {
    int o = tid + t * 256;
    int i = o >> 7;
    float hv = slots[(size_t)b * Kk * Dd + o];
    float uv = updp[((size_t)b * 16 + i) * Dd + (o & 127)] / rowsum[b * 16 + i];
    h_s[o]  = hv;
    h_bf[o] = (__bf16)hv;
    u_bf[o] = (__bf16)uv;
    h_bf[1024 + o]  = (__bf16)0.0f;   // zero padding rows 8..15
    u_bf[1024 + o]  = (__bf16)0.0f;
    ln_bf[1024 + o] = (__bf16)0.0f;
  }
  __syncthreads();

  v16bf au[4], ah[4];
  #pragma unroll
  for (int ks = 0; ks < 4; ++ks) {
    int kb = ks * 32 + half * 16;
    au[ks] = cat16(*(const v8bf*)&u_bf[row_a*128 + kb], *(const v8bf*)&u_bf[row_a*128 + kb + 8]);
    ah[ks] = cat16(*(const v8bf*)&h_bf[row_a*128 + kb], *(const v8bf*)&h_bf[row_a*128 + kb + 8]);
  }

  // gi = u @ W_ih^T + b_ih ; gh = h @ W_hh^T + b_hh  (24 column tiles each)
  for (int t = 0; t < 3; ++t) {
    int nt   = wave * 3 + t;
    int dcol = nt * 16 + row_a;
    v8f a1 = {0.f,0.f,0.f,0.f,0.f,0.f,0.f,0.f};
    v8f a2 = {0.f,0.f,0.f,0.f,0.f,0.f,0.f,0.f};
    #pragma unroll
    for (int ks = 0; ks < 4; ++ks) {
      int ko = ks * 32 + half * 16;
      a1 = WMMA_BF16(au[ks],
                     cat16(*(const v8bf*)&Wih_bf[dcol*128 + ko],
                           *(const v8bf*)&Wih_bf[dcol*128 + ko + 8]), a1);
      a2 = WMMA_BF16(ah[ks],
                     cat16(*(const v8bf*)&Whh_bf[dcol*128 + ko],
                           *(const v8bf*)&Whh_bf[dcol*128 + ko + 8]), a2);
    }
    if (half == 0) {        // only slot rows 0..7 are kept
      float bi = b_ih[dcol], bh = b_hh[dcol];
      #pragma unroll
      for (int e = 0; e < 8; ++e) {
        gi_s[e * 384 + dcol] = a1[e] + bi;
        gh_s[e * 384 + dcol] = a2[e] + bh;
      }
    }
  }
  __syncthreads();

  #pragma unroll
  for (int t = 0; t < 4; ++t) {            // GRU gates
    int o = tid + t * 256;
    int i = o >> 7, d = o & 127;
    float ir = gi_s[i*384 + d], iz = gi_s[i*384 + 128 + d], inn = gi_s[i*384 + 256 + d];
    float hr = gh_s[i*384 + d], hz = gh_s[i*384 + 128 + d], hnn = gh_s[i*384 + 256 + d];
    float r = 1.f / (1.f + __expf(-(ir + hr)));
    float z = 1.f / (1.f + __expf(-(iz + hz)));
    float n = tanhf(inn + r * hnn);
    hn_s[o] = (1.f - z) * n + z * h_s[o];
  }
  __syncthreads();

  {                                        // LN(hn) with g_mlp/b_mlp -> ln_bf
    int row = wave;
    float4 v = *(float4*)&hn_s[row * 128 + lane * 4];
    float s = v.x + v.y + v.z + v.w;
    for (int m = 16; m >= 1; m >>= 1) s += __shfl_xor(s, m, 32);
    float mean = s * (1.0f / 128.0f);
    float dx = v.x - mean, dy = v.y - mean, dz = v.z - mean, dw = v.w - mean;
    float q = dx*dx + dy*dy + dz*dz + dw*dw;
    for (int m = 16; m >= 1; m >>= 1) q += __shfl_xor(q, m, 32);
    float rinv = rsqrtf(q * (1.0f / 128.0f) + 1e-5f);
    int k = lane * 4;
    ln_bf[row*128 + k+0] = (__bf16)(dx * rinv * g_mlp[k+0] + b_mlp[k+0]);
    ln_bf[row*128 + k+1] = (__bf16)(dy * rinv * g_mlp[k+1] + b_mlp[k+1]);
    ln_bf[row*128 + k+2] = (__bf16)(dz * rinv * g_mlp[k+2] + b_mlp[k+2]);
    ln_bf[row*128 + k+3] = (__bf16)(dw * rinv * g_mlp[k+3] + b_mlp[k+3]);
  }
  __syncthreads();

  {                                        // mid = relu(ln @ W1^T + b1)
    v16bf al[4];
    #pragma unroll
    for (int ks = 0; ks < 4; ++ks) {
      int kb = ks * 32 + half * 16;
      al[ks] = cat16(*(const v8bf*)&ln_bf[row_a*128 + kb], *(const v8bf*)&ln_bf[row_a*128 + kb + 8]);
    }
    int dcol = wave * 16 + row_a;
    v8f a = {0.f,0.f,0.f,0.f,0.f,0.f,0.f,0.f};
    #pragma unroll
    for (int ks = 0; ks < 4; ++ks) {
      int ko = ks * 32 + half * 16;
      a = WMMA_BF16(al[ks],
                    cat16(*(const v8bf*)&W1_bf[dcol*128 + ko],
                          *(const v8bf*)&W1_bf[dcol*128 + ko + 8]), a);
    }
    float bb = b1[dcol];
    #pragma unroll
    for (int e = 0; e < 8; ++e)            // full tile: rows 8..15 garbage is OK
      mid_bf[(half * 8 + e) * 128 + dcol] = (__bf16)fmaxf(a[e] + bb, 0.f);
  }
  __syncthreads();

  {                                        // slots = hn + mid @ W2^T + b2
    v16bf am[4];
    #pragma unroll
    for (int ks = 0; ks < 4; ++ks) {
      int kb = ks * 32 + half * 16;
      am[ks] = cat16(*(const v8bf*)&mid_bf[row_a*128 + kb], *(const v8bf*)&mid_bf[row_a*128 + kb + 8]);
    }
    int dcol = wave * 16 + row_a;
    v8f a = {0.f,0.f,0.f,0.f,0.f,0.f,0.f,0.f};
    #pragma unroll
    for (int ks = 0; ks < 4; ++ks) {
      int ko = ks * 32 + half * 16;
      a = WMMA_BF16(am[ks],
                    cat16(*(const v8bf*)&W2_bf[dcol*128 + ko],
                          *(const v8bf*)&W2_bf[dcol*128 + ko + 8]), a);
    }
    if (half == 0) {
      float bb = b2[dcol];
      #pragma unroll
      for (int e = 0; e < 8; ++e)
        slots[(size_t)b * Kk * Dd + e * 128 + dcol] = hn_s[e * 128 + dcol] + a[e] + bb;
    }
  }
}

// ---------------------------------------------------------------------------
extern "C" void kernel_launch(void* const* d_in, const int* in_sizes, int n_in,
                              void* d_out, int out_size, void* d_ws, size_t ws_size,
                              hipStream_t stream) {
  (void)in_sizes; (void)n_in; (void)ws_size;
  const float* x        = (const float*)d_in[0];
  const float* slots_in = (const float*)d_in[1];
  const float* Wq   = (const float*)d_in[2];  const float* bq   = (const float*)d_in[3];
  const float* Wk   = (const float*)d_in[4];  const float* bk   = (const float*)d_in[5];
  const float* Wv   = (const float*)d_in[6];  const float* bv   = (const float*)d_in[7];
  const float* g_in = (const float*)d_in[8];  const float* b_in = (const float*)d_in[9];
  const float* g_sl = (const float*)d_in[10]; const float* b_sl = (const float*)d_in[11];
  const float* g_ml = (const float*)d_in[12]; const float* b_ml = (const float*)d_in[13];
  const float* W1   = (const float*)d_in[14]; const float* b1   = (const float*)d_in[15];
  const float* W2   = (const float*)d_in[16]; const float* b2   = (const float*)d_in[17];
  const float* W_ih = (const float*)d_in[18]; const float* W_hh = (const float*)d_in[19];
  const float* b_ih = (const float*)d_in[20]; const float* b_hh = (const float*)d_in[21];

  char* ws = (char*)d_ws;
  __bf16* keys   = (__bf16*)ws;  ws += (size_t)Bb * Nn * Dd * 2;   // 64 MB
  __bf16* valsT  = (__bf16*)ws;  ws += (size_t)Bb * Nn * Dd * 2;   // 64 MB
  float*  slots  = (float*)ws;   ws += (size_t)Bb * Kk * Dd * 4;
  float*  updp   = (float*)ws;   ws += (size_t)Bb * 16 * Dd * 4;
  float*  rsums  = (float*)ws;   ws += (size_t)Bb * 16 * 4;        // contiguous after updp
  __bf16* Wq_bf  = (__bf16*)ws;  ws += (size_t)Dd * Dd * 2;
  __bf16* Wih_bf = (__bf16*)ws;  ws += (size_t)3 * Dd * Dd * 2;
  __bf16* Whh_bf = (__bf16*)ws;  ws += (size_t)3 * Dd * Dd * 2;
  __bf16* W1_bf  = (__bf16*)ws;  ws += (size_t)Dd * Dd * 2;
  __bf16* W2_bf  = (__bf16*)ws;  ws += (size_t)Dd * Dd * 2;

  hipMemcpyAsync(slots, slots_in, (size_t)Bb * Kk * Dd * sizeof(float),
                 hipMemcpyDeviceToDevice, stream);

  // one-time bf16 weight conversion for the WMMA paths
  cvtbf_kernel<<<(Dd*Dd + 255)/256, 256, 0, stream>>>(Wq, Wq_bf, Dd*Dd);
  cvtbf_kernel<<<(3*Dd*Dd + 255)/256, 256, 0, stream>>>(W_ih, Wih_bf, 3*Dd*Dd);
  cvtbf_kernel<<<(3*Dd*Dd + 255)/256, 256, 0, stream>>>(W_hh, Whh_bf, 3*Dd*Dd);
  cvtbf_kernel<<<(Dd*Dd + 255)/256, 256, 0, stream>>>(W1, W1_bf, Dd*Dd);
  cvtbf_kernel<<<(Dd*Dd + 255)/256, 256, 0, stream>>>(W2, W2_bf, Dd*Dd);

  ln_kv_kernel<<<dim3(Nn / 64, Bb), 256, 0, stream>>>(
      x, Wk, bk, Wv, bv, g_in, b_in, keys, valsT);

  const int nzero = Bb * 16 * Dd + Bb * 16;   // updp + rsums (contiguous)
  for (int it = 0; it < 3; ++it) {
    zero_kernel<<<(nzero + 255) / 256, 256, 0, stream>>>(updp, nzero);
    attn_kernel<<<dim3(8, Bb), 256, 0, stream>>>(
        slots, keys, valsT, Wq_bf, bq, g_sl, b_sl, updp, rsums);
    gru_mlp_kernel<<<Bb, 256, 0, stream>>>(
        slots, updp, rsums, Wih_bf, Whh_bf, b_ih, b_hh, g_ml, b_ml,
        W1_bf, b1, W2_bf, b2);
  }

  hipMemcpyAsync(d_out, slots, (size_t)out_size * sizeof(float),
                 hipMemcpyDeviceToDevice, stream);
}
